// CustomLSTM_86904368268036
// MI455X (gfx1250) — compile-verified, matrix operand order checked by reading, not run on hardware
//
#include <hip/hip_runtime.h>
#include <hip/hip_bf16.h>

typedef _Float16 h8   __attribute__((ext_vector_type(8)));
typedef _Float16 h16v __attribute__((ext_vector_type(16)));
typedef float    v8f  __attribute__((ext_vector_type(8)));

#define T_STEPS 1024
#define BATCH   64
#define ISZ     256
#define HSZ     512
#define KTOT    768            // ISZ + HSZ (fused [x|h] GEMM)
#define NWG     32             // persistent workgroups
#define HC      16             // hidden units per WG
#define GC      64             // gate columns per WG (4*HC)
#define WP      (KTOT + 8)     // padded LDS row stride (halves) -> bank spread
#define AP      (KTOT + 8)
#define GP      (GC + 4)       // gates staging stride (floats)

// ---- workspace byte offsets ----
#define WS_W16   0u                      // 2048*768 f16            = 3,145,728
#define WS_BIAS  3145728u                // 2048 f32                = 8,192
#define WS_X16   3153920u                // 1024*64*256 f16         = 33,554,432
#define WS_HBUF  36708352u               // 2 * 64*512 f16          = 131,072
#define WS_CST   36839424u               // 64*512 f32              = 131,072
#define WS_SYNC  36970496u               // 2 ints
#define SMEM_BYTES ((GC*WP + BATCH*AP) * 2)   // 198,656 B < 320 KB/WGP

// Build W = [Wi_x | Wi_h + Wh] in f16, row-major (2048 x 768)
__global__ void prep_weights_k(const float* __restrict__ Wi,
                               const float* __restrict__ Wh,
                               _Float16* __restrict__ W16) {
  int g = blockIdx.x * blockDim.x + threadIdx.x;
  if (g >= 2048 * KTOT) return;
  int row = g / KTOT, k = g % KTOT;
  float v = Wi[g];
  if (k >= ISZ) v += Wh[row * HSZ + (k - ISZ)];
  W16[g] = (_Float16)v;
}

__global__ void prep_bias_k(const float* __restrict__ bi,
                            const float* __restrict__ bh,
                            float* __restrict__ bias) {
  int g = blockIdx.x * blockDim.x + threadIdx.x;
  if (g < 2048) bias[g] = bi[g] + bh[g];
}

// x (B,T,I) f32  ->  x16 (T,B,I) f16  (t-major so each step's slice is contiguous)
__global__ void prep_x_k(const float* __restrict__ x, _Float16* __restrict__ x16) {
  long gid = (long)blockIdx.x * blockDim.x + threadIdx.x;
  if (gid >= (long)BATCH * T_STEPS * ISZ) return;
  int i = (int)(gid % ISZ);
  long r = gid / ISZ;
  int t = (int)(r % T_STEPS);
  int b = (int)(r / T_STEPS);
  x16[((long)t * BATCH + b) * ISZ + i] = (_Float16)x[gid];
}

__global__ void init_state_k(_Float16* __restrict__ hbuf, float* __restrict__ cst,
                             int* __restrict__ sync) {
  int gid = blockIdx.x * blockDim.x + threadIdx.x;
  if (gid < 2 * BATCH * HSZ) hbuf[gid] = (_Float16)0.f;
  if (gid < BATCH * HSZ)     cst[gid]  = 0.f;
  if (gid < 2)               sync[gid] = 0;
}

__device__ inline void grid_barrier(int* cnt, int* gen, int nwg) {
  __threadfence();          // make this block's global writes (h, out) visible
  __syncthreads();
  if (threadIdx.x == 0) {
    int g = __hip_atomic_load(gen, __ATOMIC_RELAXED, __HIP_MEMORY_SCOPE_AGENT);
    int prev = __hip_atomic_fetch_add(cnt, 1, __ATOMIC_ACQ_REL, __HIP_MEMORY_SCOPE_AGENT);
    if (prev == nwg - 1) {
      __hip_atomic_store(cnt, 0, __ATOMIC_RELAXED, __HIP_MEMORY_SCOPE_AGENT);
      __hip_atomic_fetch_add(gen, 1, __ATOMIC_RELEASE, __HIP_MEMORY_SCOPE_AGENT);
    } else {
      while (__hip_atomic_load(gen, __ATOMIC_ACQUIRE, __HIP_MEMORY_SCOPE_AGENT) == g)
        __builtin_amdgcn_s_sleep(1);
    }
  }
  __syncthreads();
}

__global__ __launch_bounds__(256, 1)
void lstm_persistent_k(const _Float16* __restrict__ W16,
                       const float* __restrict__ bias,
                       const _Float16* __restrict__ x16,
                       _Float16* __restrict__ hbuf,   // double-buffered h (f16)
                       float* __restrict__ cst,       // fp32 cell state
                       int* __restrict__ sync,        // {count, generation}
                       const float* __restrict__ eta_s,
                       const float* __restrict__ eta_t,
                       float* __restrict__ out) {
  extern __shared__ char smem[];
  _Float16* Wlds = (_Float16*)smem;                          // 64 x WP  (weights, resident)
  _Float16* Alds = (_Float16*)(smem + GC * WP * 2);          // 64 x AP  ([x|h] activations)
  float*    Glds = (float*)Alds;                             // 64 x GP  (gates, aliases Alds)

  const int tid  = threadIdx.x;
  const int wave = tid >> 5;
  const int lane = tid & 31;
  const int h0   = blockIdx.x * HC;

  // ---- load this WG's weight slice into LDS once (stays for all 1024 steps) ----
  for (int it = 0; it < 24; ++it) {
    int c  = tid + 256 * it;           // 64 rows * 96 8-half chunks = 6144
    int lc = c / 96;                   // local gate col 0..63
    int kc = (c % 96) * 8;
    int grow = (lc >> 4) * HSZ + h0 + (lc & 15);   // gate*512 + h0 + within
    *(h8*)(Wlds + lc * WP + kc) = *(const h8*)(W16 + (long)grow * KTOT + kc);
  }

  const float es0 = eta_s[0], es1 = eta_s[1], es2 = eta_s[2], es3 = eta_s[3];
  const float et0 = eta_t[0], et1 = eta_t[1], et2 = eta_t[2], et3 = eta_t[3];

  // tile assignment: wave -> 1 col tile (16 gate cols) x 2 row tiles (32 batches)
  const int cb  = (wave & 3) * 16;
  const int mt0 = (wave >> 2) * 2;
  const int r   = lane & 15;
  const int hiK = (lane >> 4) * 8;     // ISA A/B K interleave: {0-7,16-23} / {8-15,24-31}

  float* hTp = out + (long)BATCH * T_STEPS * HSZ;
  float* cTp = hTp + BATCH * HSZ;

  for (int t = 0; t < T_STEPS; ++t) {
    const _Float16* hread  = hbuf + (t & 1) * (BATCH * HSZ);
    _Float16*       hwrite = hbuf + ((t + 1) & 1) * (BATCH * HSZ);
    const _Float16* xrow   = x16 + (long)t * BATCH * ISZ;

    // ---- stage A = [x_t | h] (64 x 768 f16) into LDS ----
    for (int it = 0; it < 24; ++it) {
      int c  = tid + 256 * it;
      int b  = c / 96;
      int kc = (c % 96) * 8;
      h8 v;
      if (kc < ISZ) v = *(const h8*)(xrow + b * ISZ + kc);
      else          v = *(const h8*)(hread + b * HSZ + (kc - ISZ));
      *(h8*)(Alds + b * AP + kc) = v;
    }
    __syncthreads();

    // ---- GEMM: gates_tile = A x W^T via v_wmma_f32_16x16x32_f16 ----
    v8f acc0 = {}; v8f acc1 = {};
    const _Float16* Bp  = Wlds + (cb + r) * WP + hiK;        // lane = column of W
    const _Float16* Ap0 = Alds + (mt0 * 16 + r) * AP + hiK;  // lane = row of A
    const _Float16* Ap1 = Ap0 + 16 * AP;
    #pragma unroll 4
    for (int k = 0; k < KTOT; k += 32) {
      h16v bf = __builtin_shufflevector(*(const h8*)(Bp + k), *(const h8*)(Bp + k + 16),
                                        0,1,2,3,4,5,6,7,8,9,10,11,12,13,14,15);
      h16v a0 = __builtin_shufflevector(*(const h8*)(Ap0 + k), *(const h8*)(Ap0 + k + 16),
                                        0,1,2,3,4,5,6,7,8,9,10,11,12,13,14,15);
      h16v a1 = __builtin_shufflevector(*(const h8*)(Ap1 + k), *(const h8*)(Ap1 + k + 16),
                                        0,1,2,3,4,5,6,7,8,9,10,11,12,13,14,15);
      acc0 = __builtin_amdgcn_wmma_f32_16x16x32_f16(false, a0, false, bf, (short)0, acc0, false, false);
      acc1 = __builtin_amdgcn_wmma_f32_16x16x32_f16(false, a1, false, bf, (short)0, acc1, false, false);
    }
    __syncthreads();   // everyone done reading Alds before we alias it with gates

    // ---- spill accumulators to gates LDS: D layout M = rr + 8*(lane>=16), N = lane%16 ----
    {
      int col = cb + r;
      int b0  = mt0 * 16 + (lane >> 4) * 8;
      #pragma unroll
      for (int rr = 0; rr < 8; ++rr) {
        Glds[(b0 + rr) * GP + col]      = acc0[rr];
        Glds[(b0 + 16 + rr) * GP + col] = acc1[rr];
      }
    }
    __syncthreads();

    // ---- elementwise gates + state update: 64 batches x 16 hidden, 4 per thread ----
    #pragma unroll
    for (int u = 0; u < 4; ++u) {
      int e = tid + 256 * u;
      int b = e >> 4;
      int j = e & 15;
      float zi = Glds[b * GP +      j] + bias[          h0 + j];
      float zf = Glds[b * GP + 16 + j] + bias[  HSZ   + h0 + j];
      float zc = Glds[b * GP + 32 + j] + bias[2 * HSZ + h0 + j];
      float zo = Glds[b * GP + 48 + j] + bias[3 * HSZ + h0 + j];
      float ig = es0 + es1 / (1.f + __expf(-(zi - es2) * es3));
      float fg = es0 + es1 / (1.f + __expf(-(zf - es2) * es3));
      float cg = et0 + et1 * tanhf((zc - et2) * et3);
      float og = es0 + es1 / (1.f + __expf(-(zo - es2) * es3));
      int hidx = b * HSZ + h0 + j;
      float cn = fg * cst[hidx] + ig * cg;
      cst[hidx] = cn;
      float hn = og * (et0 + et1 * tanhf((cn - et2) * et3));
      out[(long)b * T_STEPS * HSZ + (long)t * HSZ + h0 + j] = hn;
      hwrite[hidx] = (_Float16)hn;
      if (t == T_STEPS - 1) { hTp[hidx] = hn; cTp[hidx] = cn; }
    }

    grid_barrier(sync, sync + 1, NWG);   // h double-buffer handoff
  }
}

extern "C" void kernel_launch(void* const* d_in, const int* in_sizes, int n_in,
                              void* d_out, int out_size, void* d_ws, size_t ws_size,
                              hipStream_t stream) {
  (void)in_sizes; (void)n_in; (void)out_size; (void)ws_size;
  const float* x     = (const float*)d_in[0];
  const float* Wi    = (const float*)d_in[1];
  const float* bi    = (const float*)d_in[2];
  const float* Wh    = (const float*)d_in[3];
  const float* bh    = (const float*)d_in[4];
  const float* eta_s = (const float*)d_in[5];
  const float* eta_t = (const float*)d_in[6];
  float* out = (float*)d_out;

  char* ws = (char*)d_ws;
  _Float16* W16  = (_Float16*)(ws + WS_W16);
  float*    bias = (float*)   (ws + WS_BIAS);
  _Float16* x16  = (_Float16*)(ws + WS_X16);
  _Float16* hbuf = (_Float16*)(ws + WS_HBUF);
  float*    cst  = (float*)   (ws + WS_CST);
  int*      sync = (int*)     (ws + WS_SYNC);

  prep_weights_k<<<(2048 * KTOT) / 256, 256, 0, stream>>>(Wi, Wh, W16);
  prep_bias_k<<<8, 256, 0, stream>>>(bi, bh, bias);
  prep_x_k<<<(BATCH * T_STEPS * ISZ) / 256, 256, 0, stream>>>(x, x16);
  init_state_k<<<(2 * BATCH * HSZ) / 256, 256, 0, stream>>>(hbuf, cst, sync);
  lstm_persistent_k<<<NWG, 256, SMEM_BYTES, stream>>>(W16, bias, x16, hbuf, cst, sync,
                                                      eta_s, eta_t, out);
}
